// Head_82738249990240
// MI455X (gfx1250) — matrix operand order repaired and checked by reading, not compiled, not run on hardware
//
#include <hip/hip_runtime.h>

// Problem dims (fixed by reference)
#define Bn 32
#define Sn 1024
#define En 768
#define Dn 64
#define NSLAB (Sn * Dn)   // 65536 elements per LN slab

typedef __attribute__((ext_vector_type(16))) _Float16 v16h;
typedef __attribute__((ext_vector_type(8)))  _Float16 v8h;
typedef __attribute__((ext_vector_type(8)))  float    v8f;
typedef __attribute__((ext_vector_type(4)))  float    v4f;

__device__ __forceinline__ v16h pack2(v8h a, v8h b) {
  v16h f;
#pragma unroll
  for (int i = 0; i < 8; ++i) { f[i] = a[i]; f[8 + i] = b[i]; }
  return f;
}

// Async global->LDS copy of 16 bytes per lane (ASYNCcnt-tracked).
__device__ __forceinline__ void async_cp16(uint32_t lds_off, const void* gptr) {
  asm volatile("global_load_async_to_lds_b128 %0, %1, off"
               :: "v"(lds_off), "v"(gptr) : "memory");
}
__device__ __forceinline__ uint32_t lds_off32(const void* p) {
  return (uint32_t)(uintptr_t)p;   // LDS aperture: addr[31:0] is the LDS offset
}

// ---------------------------------------------------------------------------
// k_prep: Wt[m][n][e] = (f16) W_m[e][n]   (column-major weights for B-frags)
// ---------------------------------------------------------------------------
__global__ void __launch_bounds__(256) k_prep(const float* __restrict__ Wq,
                                              const float* __restrict__ Wk,
                                              const float* __restrict__ Wv,
                                              _Float16* __restrict__ Wt) {
  int idx = blockIdx.x * 256 + threadIdx.x;      // grid sized exactly 3*64*768
  int m = idx / (Dn * En);
  int r = idx - m * (Dn * En);
  int n = r / En;
  int e = r - n * En;
  const float* W = (m == 0) ? Wq : ((m == 1) ? Wk : Wv);
  Wt[idx] = (_Float16)W[e * Dn + n];
}

// ---------------------------------------------------------------------------
// k_proj: Q/K/V = x @ W via f16 WMMA, f32 accumulate; also LN partial stats.
// One wave = 16 rows of x; block = 8 waves = 128 rows (within one batch).
// ---------------------------------------------------------------------------
__global__ void __launch_bounds__(256) k_proj(const float* __restrict__ x,
                                              const _Float16* __restrict__ Wt,
                                              _Float16* __restrict__ Qh,
                                              _Float16* __restrict__ Kh,
                                              _Float16* __restrict__ Vh,
                                              float* __restrict__ stats) {
  __shared__ float sred[8][3][2];

  const int lane = threadIdx.x & 31;
  const int wave = threadIdx.x >> 5;
  const int hi = lane >> 4;       // lane half
  const int ln = lane & 15;
  const long rowbase = (long)blockIdx.x * 128 + wave * 16;
  const float* xrow = x + (rowbase + ln) * (long)En;

  v8f acc[3][4];
  v8f z = {};
#pragma unroll
  for (int m = 0; m < 3; ++m)
#pragma unroll
    for (int t = 0; t < 4; ++t) acc[m][t] = z;

  for (int kc = 0; kc < En; kc += 32) {
    // A fragment of x (16x32, f16): lane row = ln, K split per A layout.
    v4f a0 = *(const v4f*)(xrow + kc + hi * 8);
    v4f a1 = *(const v4f*)(xrow + kc + hi * 8 + 4);
    v4f a2 = *(const v4f*)(xrow + kc + 16 + hi * 8);
    v4f a3 = *(const v4f*)(xrow + kc + 16 + hi * 8 + 4);
    v16h af;
#pragma unroll
    for (int i = 0; i < 4; ++i) {
      af[i]      = (_Float16)a0[i];
      af[4 + i]  = (_Float16)a1[i];
      af[8 + i]  = (_Float16)a2[i];
      af[12 + i] = (_Float16)a3[i];
    }
#pragma unroll
    for (int m = 0; m < 3; ++m) {
#pragma unroll
      for (int t = 0; t < 4; ++t) {
        // B fragment (32x16): lane col n = t*16+ln; K = 8*hi+s, then 16+8*hi+s
        const _Float16* colp =
            Wt + ((long)m * Dn + t * 16 + ln) * En + kc + hi * 8;
        v8h b0 = *(const v8h*)colp;
        v8h b1 = *(const v8h*)(colp + 16);
        acc[m][t] = __builtin_amdgcn_wmma_f32_16x16x32_f16(
            false, af, false, pack2(b0, b1), (short)0, acc[m][t], false, false);
      }
    }
  }

  // Store f16 outputs + per-batch sum / sumsq for the whole-slab LayerNorm.
#pragma unroll
  for (int m = 0; m < 3; ++m) {
    _Float16* op = (m == 0) ? Qh : ((m == 1) ? Kh : Vh);
    float s = 0.f, q = 0.f;
#pragma unroll
    for (int t = 0; t < 4; ++t)
#pragma unroll
      for (int r = 0; r < 8; ++r) {
        float v = acc[m][t][r];
        s += v; q += v * v;
        op[(rowbase + r + 8 * hi) * (long)Dn + t * 16 + ln] = (_Float16)v;
      }
#pragma unroll
    for (int off = 16; off >= 1; off >>= 1) {
      s += __shfl_xor(s, off, 32);
      q += __shfl_xor(q, off, 32);
    }
    if (lane == 0) { sred[wave][m][0] = s; sred[wave][m][1] = q; }
  }
  __syncthreads();
  if (threadIdx.x < 6) {
    int m = threadIdx.x >> 1, j = threadIdx.x & 1;
    float t = 0.f;
#pragma unroll
    for (int w = 0; w < 8; ++w) t += sred[w][m][j];
    int batch = (blockIdx.x * 128) / Sn;
    atomicAdd(stats + (m * Bn + batch) * 2 + j, t);
  }
}

// ---------------------------------------------------------------------------
// k_norm: apply LayerNorm. Q gets the 1/sqrt(d) score scale folded in.
// V is written transposed [d][S] so PV B-fragments are contiguous.
// ---------------------------------------------------------------------------
__global__ void __launch_bounds__(256) k_norm(_Float16* __restrict__ Qh,
                                              _Float16* __restrict__ Kh,
                                              const _Float16* __restrict__ Vh,
                                              _Float16* __restrict__ Vt,
                                              const float* __restrict__ stats) {
  long idx = (long)blockIdx.x * 256 + threadIdx.x;   // over Bn*Dn*Sn
  long b = idx / ((long)Dn * Sn);
  long r2 = idx - b * ((long)Dn * Sn);
  int n = (int)(r2 / Sn);
  int s = (int)(r2 - (long)n * Sn);

  float mu[3], rs[3];
#pragma unroll
  for (int m = 0; m < 3; ++m) {
    float su = stats[(m * Bn + b) * 2 + 0];
    float sq = stats[(m * Bn + b) * 2 + 1];
    float mean = su * (1.0f / NSLAB);
    float var = sq * (1.0f / NSLAB) - mean * mean;
    mu[m] = mean;
    rs[m] = rsqrtf(var + 1e-5f);
  }
  long base = b * (long)(Sn * Dn);
  long qi = base + r2;
  Qh[qi] = (_Float16)(((float)Qh[qi] - mu[0]) * rs[0] * 0.125f);  // 1/sqrt(64)
  Kh[qi] = (_Float16)(((float)Kh[qi] - mu[1]) * rs[1]);
  float v = (float)Vh[base + (long)s * Dn + n];
  Vt[idx] = (_Float16)((v - mu[2]) * rs[2]);
}

// ---------------------------------------------------------------------------
// k_attn: flash attention with double-buffered async LDS staging of K/V.
// Wave = 16 query rows; block = 128 rows; key tile = 64 keys.
// ---------------------------------------------------------------------------
#define KT 64
#define KROW 72   // padded LDS row (halves): 144 B, 16B-aligned

__global__ void __launch_bounds__(256) k_attn(const _Float16* __restrict__ Qh,
                                              const _Float16* __restrict__ Kh,
                                              const _Float16* __restrict__ Vt,
                                              float* __restrict__ out) {
  __shared__ _Float16 kbuf[2][KT][KROW];   // [key][dim], 18 KB
  __shared__ _Float16 vbuf[2][Dn][KROW];   // [dim][key], 18 KB
  __shared__ _Float16 pbuf[8][16][KROW];   // per-wave P tile, 18 KB

  const int lane = threadIdx.x & 31;
  const int wave = threadIdx.x >> 5;
  const int hi = lane >> 4;
  const int ln = lane & 15;
  const int b = blockIdx.x >> 3;
  const int qblk = blockIdx.x & 7;
  const long qrow0 = (long)b * Sn + qblk * 128 + wave * 16;

  // Q A-fragments (dims 0..31 and 32..63), already scaled by 1/sqrt(d).
  const _Float16* qp = Qh + (qrow0 + ln) * (long)Dn;
  v16h qf0 = pack2(*(const v8h*)(qp + hi * 8), *(const v8h*)(qp + 16 + hi * 8));
  v16h qf1 = pack2(*(const v8h*)(qp + 32 + hi * 8), *(const v8h*)(qp + 48 + hi * 8));

  const _Float16* kb = Kh + (long)b * Sn * Dn;  // [key][dim]
  const _Float16* vb = Vt + (long)b * Dn * Sn;  // [dim][key]

  // Cooperative async stage of one 64-key tile (4 async instrs per wave).
  const int cr = threadIdx.x >> 3;   // 0..31
  const int cc = threadIdx.x & 7;    // 16B chunk
  auto stage = [&](int kt, int bufi) {
    async_cp16(lds_off32(&kbuf[bufi][cr][cc * 8]),
               (const char*)(kb + (long)(kt + cr) * Dn) + cc * 16);
    async_cp16(lds_off32(&kbuf[bufi][cr + 32][cc * 8]),
               (const char*)(kb + (long)(kt + cr + 32) * Dn) + cc * 16);
    async_cp16(lds_off32(&vbuf[bufi][cr][cc * 8]),
               (const char*)(vb + (long)cr * Sn + kt) + cc * 16);
    async_cp16(lds_off32(&vbuf[bufi][cr + 32][cc * 8]),
               (const char*)(vb + (long)(cr + 32) * Sn + kt) + cc * 16);
  };

  v8f acc[4];
  v8f z = {};
#pragma unroll
  for (int t = 0; t < 4; ++t) acc[t] = z;
  float mrow[8], lrow[8];
#pragma unroll
  for (int r = 0; r < 8; ++r) { mrow[r] = -__builtin_inff(); lrow[r] = 0.f; }

  stage(0, 0);
  for (int it = 0; it < Sn / KT; ++it) {
    const int buf = it & 1;
    if (it + 1 < Sn / KT) {
      stage((it + 1) * KT, buf ^ 1);
      asm volatile("s_wait_asynccnt 0x4" ::: "memory");  // tile `it` arrived
    } else {
      asm volatile("s_wait_asynccnt 0x0" ::: "memory");
    }
    __syncthreads();

    // --- scores: 4 N-tiles of 16 keys, K-dim 64 in two chunks ---
    v8f sc[4];
#pragma unroll
    for (int j = 0; j < 4; ++j) {
      sc[j] = z;
      const _Float16* cp = &kbuf[buf][j * 16 + ln][0];
      sc[j] = __builtin_amdgcn_wmma_f32_16x16x32_f16(false, qf0, false,
          pack2(*(const v8h*)(cp + hi * 8), *(const v8h*)(cp + 16 + hi * 8)),
          (short)0, sc[j], false, false);
      sc[j] = __builtin_amdgcn_wmma_f32_16x16x32_f16(false, qf1, false,
          pack2(*(const v8h*)(cp + 32 + hi * 8), *(const v8h*)(cp + 48 + hi * 8)),
          (short)0, sc[j], false, false);
    }

    // --- online softmax (row = r + 8*hi lives in this lane half) ---
    float scale[8];
#pragma unroll
    for (int r = 0; r < 8; ++r) {
      float t = fmaxf(fmaxf(sc[0][r], sc[1][r]), fmaxf(sc[2][r], sc[3][r]));
#pragma unroll
      for (int off = 8; off >= 1; off >>= 1) t = fmaxf(t, __shfl_xor(t, off, 16));
      float mn = fmaxf(mrow[r], t);
      scale[r] = __expf(mrow[r] - mn);
      mrow[r] = mn;
      float rsum = 0.f;
#pragma unroll
      for (int j = 0; j < 4; ++j) {
        float p = __expf(sc[j][r] - mn);
        sc[j][r] = p;
        rsum += p;
      }
#pragma unroll
      for (int off = 8; off >= 1; off >>= 1) rsum += __shfl_xor(rsum, off, 16);
      lrow[r] = lrow[r] * scale[r] + rsum;
    }
#pragma unroll
    for (int t = 0; t < 4; ++t)
#pragma unroll
      for (int r = 0; r < 8; ++r) acc[t][r] *= scale[r];

    // --- C-layout -> A-layout via wave-private LDS ---
#pragma unroll
    for (int j = 0; j < 4; ++j)
#pragma unroll
      for (int r = 0; r < 8; ++r)
        pbuf[wave][r + 8 * hi][j * 16 + ln] = (_Float16)sc[j][r];
    v16h ap0 = pack2(*(const v8h*)&pbuf[wave][ln][hi * 8],
                     *(const v8h*)&pbuf[wave][ln][16 + hi * 8]);
    v16h ap1 = pack2(*(const v8h*)&pbuf[wave][ln][32 + hi * 8],
                     *(const v8h*)&pbuf[wave][ln][48 + hi * 8]);

    // --- PV: acc += P(16x64) · V(64x64) ---
#pragma unroll
    for (int t = 0; t < 4; ++t) {
      const _Float16* vp = &vbuf[buf][t * 16 + ln][0];
      acc[t] = __builtin_amdgcn_wmma_f32_16x16x32_f16(false, ap0, false,
          pack2(*(const v8h*)(vp + hi * 8), *(const v8h*)(vp + 16 + hi * 8)),
          (short)0, acc[t], false, false);
      acc[t] = __builtin_amdgcn_wmma_f32_16x16x32_f16(false, ap1, false,
          pack2(*(const v8h*)(vp + 32 + hi * 8), *(const v8h*)(vp + 48 + hi * 8)),
          (short)0, acc[t], false, false);
    }
    __syncthreads();   // all waves done with buf before it is re-staged
  }

  // Epilogue: out = acc / l
#pragma unroll
  for (int t = 0; t < 4; ++t)
#pragma unroll
    for (int r = 0; r < 8; ++r) {
      long row = qrow0 + r + 8 * hi;
      out[row * Dn + t * 16 + ln] = acc[t][r] * (1.0f / lrow[r]);
    }
}

// ---------------------------------------------------------------------------
extern "C" void kernel_launch(void* const* d_in, const int* in_sizes, int n_in,
                              void* d_out, int out_size, void* d_ws, size_t ws_size,
                              hipStream_t stream) {
  const float* x  = (const float*)d_in[0];
  const float* Wq = (const float*)d_in[1];
  const float* Wk = (const float*)d_in[2];
  const float* Wv = (const float*)d_in[3];
  float* out = (float*)d_out;

  char* ws = (char*)d_ws;
  float* stats = (float*)(ws);                     //  768 B (3*32*2 floats)
  _Float16* Wt = (_Float16*)(ws + 1024);           //  294912 B
  _Float16* Qh = (_Float16*)(ws + 1024 + 294912);  //  4 MB each below
  _Float16* Kh = Qh + (long)Bn * Sn * Dn;
  _Float16* Vh = Kh + (long)Bn * Sn * Dn;
  _Float16* Vt = Vh + (long)Bn * Sn * Dn;

  hipMemsetAsync(stats, 0, 192 * sizeof(float), stream);
  k_prep<<<(3 * Dn * En) / 256, 256, 0, stream>>>(Wq, Wk, Wv, Wt);
  k_proj<<<(Bn * Sn) / 128, 256, 0, stream>>>(x, Wt, Qh, Kh, Vh, stats);
  k_norm<<<((long)Bn * Dn * Sn) / 256, 256, 0, stream>>>(Qh, Kh, Vh, Vt, stats);
  k_attn<<<(Bn * Sn) / 128, 256, 0, stream>>>(Qh, Kh, Vt, out);
}
